// MultiHeadAttention_27762668602001
// MI455X (gfx1250) — matrix operand order repaired and checked by reading, not compiled, not run on hardware
//
#include <hip/hip_runtime.h>
#include <hip/hip_bf16.h>
#include <math.h>
#include <stdint.h>

typedef __attribute__((ext_vector_type(16))) _Float16 v16h;
typedef __attribute__((ext_vector_type(8)))  _Float16 v8h;
typedef __attribute__((ext_vector_type(4)))  _Float16 v4h;
typedef __attribute__((ext_vector_type(8)))  float    v8f;

#define EMBED 1024
#define HEADS 16
#define HDIM  64
#define BATCH 4
#define SEQ   2048
#define ROWS  (BATCH * SEQ)   // 8192

// ---------------------------------------------------------------------------
// CDNA5 async copy (global -> LDS), tracked by ASYNCcnt.  No portable builtin
// on this toolchain, so use the ISA mnemonics directly (CDNA5 ISA).
// Each lane copies 16 bytes: LDS[vdst_lane] = MEM[vaddr_lane].
// ---------------------------------------------------------------------------
static __device__ __forceinline__ void async_load_b128(unsigned lds_off,
                                                       const void* g) {
  asm volatile("global_load_async_to_lds_b128 %0, %1, off"
               :: "v"(lds_off), "v"((unsigned long long)(uintptr_t)g)
               : "memory");
}
#define WAIT_ASYNC(n) asm volatile("s_wait_asynccnt %0" :: "i"(n) : "memory")

static __device__ __forceinline__ unsigned lds_off(const void* p) {
  return (unsigned)(uintptr_t)p;   // LDS aperture: addr[31:0] = LDS offset
}

// ---------------------------------------------------------------------------
// Fragment layouts used here (wave32 WMMA 16x16x32 f16, contiguous k order for A/B, not the ISA table order)
// A (16x32, MxK): lane L -> row m = L%16; elem e -> K = e + 16*(L/16)
// B (32x16, KxN): lane L -> col n = L%16; elem e -> K = e + 16*(L/16)
// C/D (16x16 f32): lane L -> col n = L%16; VGPR r -> row m = r + 8*(L/16)
// ---------------------------------------------------------------------------
static __device__ __forceinline__ v16h pack_frag(v8h lo, v8h hi) {
  union { v16h f; v8h h[2]; } u;
  u.h[0] = lo; u.h[1] = hi;
  return u.f;
}

// A-fragment from a row-major [*, ld] f16 matrix (global or LDS)
template <typename P>
static __device__ __forceinline__ v16h load_a_frag(P base, int ld, int row,
                                                   int k, int lane) {
  int m  = row + (lane & 15);
  int ko = k + ((lane >> 4) << 4);
  return pack_frag(*(const v8h*)(base + (size_t)m * ld + ko),
                   *(const v8h*)(base + (size_t)m * ld + ko + 8));
}

// B-fragment from Bt row-major [N, K] (column n of B = contiguous K-run)
template <typename P>
static __device__ __forceinline__ v16h load_b_frag(P bt, int ld, int col,
                                                   int k, int lane) {
  int n  = col + (lane & 15);
  int ko = k + ((lane >> 4) << 4);
  return pack_frag(*(const v8h*)(bt + (size_t)n * ld + ko),
                   *(const v8h*)(bt + (size_t)n * ld + ko + 8));
}

static __device__ __forceinline__ v8f wmma_f16(v16h a, v16h b, v8f c) {
  return __builtin_amdgcn_wmma_f32_16x16x32_f16(false, a, false, b,
                                                (short)0, c, false, false);
}

// ---------------------------------------------------------------------------
// fp32 -> f16 conversion, 4-wide
// ---------------------------------------------------------------------------
__global__ void cvt_f32_f16(const float* __restrict__ in,
                            _Float16* __restrict__ out, int n4) {
  int i = blockIdx.x * blockDim.x + threadIdx.x;
  if (i < n4) {
    float4 v = ((const float4*)in)[i];
    v4h o;
    o[0] = (_Float16)v.x; o[1] = (_Float16)v.y;
    o[2] = (_Float16)v.z; o[3] = (_Float16)v.w;
    ((v4h*)out)[i] = o;
  }
}

// ---------------------------------------------------------------------------
// f16 GEMM with async-to-LDS double buffering.
// C[M,N] = A[M,K] @ Bt[N,K]^T + bias
// block = 256 threads = 8 waves (2 M x 4 N), wave tile 32x64, block 64x256.
// Per K-step the block async-stages A(64x32)=4KB + B(256x32)=16KB into LDS
// (5 async b128 per thread), waves consume via ds_load_b128.
// MODE 0: fp32 out.  MODE 1: QKV epilogue -> scatter Q/K/Vt f16.
// ---------------------------------------------------------------------------
template <int MODE>
__global__ __launch_bounds__(256) void gemm_f16(
    const _Float16* __restrict__ A, const _Float16* __restrict__ Bt,
    const float* __restrict__ bias, float* __restrict__ out,
    _Float16* __restrict__ qh, _Float16* __restrict__ kh,
    _Float16* __restrict__ vth, int Nout, int K) {
  __shared__ __align__(16) _Float16 As[2][64 * 32];    //  4 KB per buffer
  __shared__ __align__(16) _Float16 Bs[2][256 * 32];   // 16 KB per buffer

  const int tid  = threadIdx.x;
  const int lane = tid & 31;
  const int w    = tid >> 5;
  const int mblk = blockIdx.y * 64;
  const int nblk = blockIdx.x * 256;
  const int wmof = (w & 1) * 32;    // wave M offset inside block tile
  const int wnof = (w >> 1) * 64;   // wave N offset inside block tile

  // cooperative async stage of one 32-wide K slice into buffer `buf`
  auto issue_tile = [&](int k, int buf) {
    {  // A tile: 64 rows x 32 halves, 4 chunks/row
      int row = tid >> 2, c = tid & 3;
      async_load_b128(lds_off(&As[buf][row * 32 + c * 8]),
                      A + (size_t)(mblk + row) * K + k + c * 8);
    }
#pragma unroll
    for (int i = 0; i < 4; ++i) {  // B tile: 256 rows x 32 halves
      int idx = tid + 256 * i;
      int row = idx >> 2, c = idx & 3;
      async_load_b128(lds_off(&Bs[buf][row * 32 + c * 8]),
                      Bt + (size_t)(nblk + row) * K + k + c * 8);
    }
  };

  v8f acc[2][4];
  for (int mi = 0; mi < 2; ++mi)
    for (int ni = 0; ni < 4; ++ni)
      for (int r = 0; r < 8; ++r) acc[mi][ni][r] = 0.f;

  const int nk = K / 32;
  issue_tile(0, 0);
  for (int kk = 0; kk < nk; ++kk) {
    const int buf = kk & 1;
    if (kk + 1 < nk) {
      issue_tile((kk + 1) * 32, buf ^ 1);
      WAIT_ASYNC(5);            // 5 outstanding = next tile only
    } else {
      WAIT_ASYNC(0);
    }
    __syncthreads();            // all waves' portions of `buf` visible

    v16h a0 = load_a_frag(&As[buf][0], 32, wmof,      0, lane);
    v16h a1 = load_a_frag(&As[buf][0], 32, wmof + 16, 0, lane);
#pragma unroll
    for (int ni = 0; ni < 4; ++ni) {
      v16h b = load_b_frag(&Bs[buf][0], 32, wnof + 16 * ni, 0, lane);
      acc[0][ni] = wmma_f16(a0, b, acc[0][ni]);
      acc[1][ni] = wmma_f16(a1, b, acc[1][ni]);
    }
    __syncthreads();            // done reading `buf` before it is re-filled
  }

#pragma unroll
  for (int mi = 0; mi < 2; ++mi) {
#pragma unroll
    for (int ni = 0; ni < 4; ++ni) {
      const int col = nblk + wnof + 16 * ni + (lane & 15);
      const float bv = bias[col];
      const int mrow0 = mblk + wmof + 16 * mi + ((lane >> 4) << 3);
#pragma unroll
      for (int r = 0; r < 8; ++r) {
        const float v = acc[mi][ni][r] + bv;
        const int m = mrow0 + r;
        if (MODE == 0) {
          out[(size_t)m * Nout + col] = v;
        } else {
          // col encodes (t, h, d): t = col/1024, h = (col/64)%16, d = col%64
          const int t = col >> 10, h = (col >> 6) & 15, d = col & 63;
          const int bi = m >> 11, n = m & 2047;     // m = bi*SEQ + n
          if (t == 0) {
            qh[((size_t)(bi * HEADS + h) * SEQ + n) * HDIM + d] =
                (_Float16)(v * 0.125f);              // fold 1/sqrt(64)
          } else if (t == 1) {
            kh[((size_t)(bi * HEADS + h) * SEQ + n) * HDIM + d] = (_Float16)v;
          } else {
            vth[((size_t)(bi * HEADS + h) * HDIM + d) * SEQ + n] = (_Float16)v;
          }
        }
      }
    }
  }
}

// ---------------------------------------------------------------------------
// Flash attention: block = 8 waves sharing one (b,h); wave = 16 queries.
// K/V 32-key tiles are async-staged into LDS (4KB + 4KB, double buffered)
// and shared by all 8 waves (8x less global K/V traffic).
// S^T = K @ Q^T keeps queries on the lane axis -> softmax stats are a
// per-lane scalar (one shfl_xor(16)); P^T B-fragment rebuilt with 16
// shfl_xor(16) swaps, no LDS transpose.  O^T = V^T @ P^T.
// ---------------------------------------------------------------------------
__global__ __launch_bounds__(256) void attention_f16(
    const _Float16* __restrict__ qh, const _Float16* __restrict__ kh,
    const _Float16* __restrict__ vth, _Float16* __restrict__ attn) {
  __shared__ __align__(16) _Float16 Ks[2][32 * 64];   // 4 KB per buffer
  __shared__ __align__(16) _Float16 Vs[2][64 * 32];   // 4 KB per buffer

  const int tid  = threadIdx.x;
  const int lane = tid & 31;
  const int w    = tid >> 5;
  const int bh   = blockIdx.x >> 4;              // 0..63 = (b,h)
  const int qtile = (blockIdx.x & 15) * 8 + w;   // 128 q-tiles per (b,h)
  const int h = bh & 15, bi = bh >> 4;

  const _Float16* Q  = qh  + (size_t)bh * SEQ * HDIM;
  const _Float16* Kp = kh  + (size_t)bh * SEQ * HDIM;
  const _Float16* Vt = vth + (size_t)bh * HDIM * SEQ;
  const int qbase = qtile * 16;

  auto issue_tile = [&](int j, int buf) {
    {  // K tile: 32 rows x 64 halves, 8 chunks/row
      int row = tid >> 3, c = tid & 7;
      async_load_b128(lds_off(&Ks[buf][row * 64 + c * 8]),
                      Kp + (size_t)(j + row) * HDIM + c * 8);
    }
    {  // V^T tile: 64 rows x 32 halves (keys j..j+31), 4 chunks/row
      int row = tid >> 2, c = tid & 3;
      async_load_b128(lds_off(&Vs[buf][row * 32 + c * 8]),
                      Vt + (size_t)row * SEQ + j + c * 8);
    }
  };

  // Q^T B-fragments (cols = 16 queries, K = head dim): 2 loads, hoisted
  v16h qb[2];
#pragma unroll
  for (int kc = 0; kc < 2; ++kc)
    qb[kc] = load_b_frag(Q, HDIM, qbase, 32 * kc, lane);

  v8f o_acc[4];
  for (int t = 0; t < 4; ++t)
    for (int r = 0; r < 8; ++r) o_acc[t][r] = 0.f;
  float mrow = -3.0e38f, lsum = 0.f;
  const bool hiHalf = lane >= 16;

  const int nt = SEQ / 32;
  issue_tile(0, 0);
  for (int jt = 0; jt < nt; ++jt) {
    const int buf = jt & 1;
    if (jt + 1 < nt) {
      issue_tile((jt + 1) * 32, buf ^ 1);
      WAIT_ASYNC(2);
    } else {
      WAIT_ASYNC(0);
    }
    __syncthreads();

    // S^T tile: 32 keys x 16 queries = two 16x16 accumulators
    v8f s0, s1;
    for (int r = 0; r < 8; ++r) { s0[r] = 0.f; s1[r] = 0.f; }
#pragma unroll
    for (int kc = 0; kc < 2; ++kc) {
      v16h ka0 = load_a_frag(&Ks[buf][0], 64, 0,  32 * kc, lane);
      v16h ka1 = load_a_frag(&Ks[buf][0], 64, 16, 32 * kc, lane);
      s0 = wmma_f16(ka0, qb[kc], s0);
      s1 = wmma_f16(ka1, qb[kc], s1);
    }

    // online softmax (per-query stats live on the lane axis)
    float tmax = s0[0];
#pragma unroll
    for (int r = 0; r < 8; ++r) {
      tmax = fmaxf(tmax, s0[r]);
      tmax = fmaxf(tmax, s1[r]);
    }
    tmax = fmaxf(tmax, __shfl_xor(tmax, 16));
    const float mnew = fmaxf(mrow, tmax);
    const float scale = __expf(mrow - mnew);
    mrow = mnew;

    v8f p0, p1;
    float psum = 0.f;
#pragma unroll
    for (int r = 0; r < 8; ++r) {
      p0[r] = __expf(s0[r] - mnew);
      p1[r] = __expf(s1[r] - mnew);
      psum += p0[r] + p1[r];
    }
    psum += __shfl_xor(psum, 16);
    lsum = lsum * scale + psum;
#pragma unroll
    for (int t = 0; t < 4; ++t)
#pragma unroll
      for (int r = 0; r < 8; ++r) o_acc[t][r] *= scale;

    // P^T B-fragment: lane L -> col q = L%16, elem e -> key 16*(L/16)+e.
    v8f p0s, p1s;
#pragma unroll
    for (int r = 0; r < 8; ++r) {
      p0s[r] = __shfl_xor(p0[r], 16);
      p1s[r] = __shfl_xor(p1[r], 16);
    }
    v16h pf;
#pragma unroll
    for (int e = 0; e < 8; ++e) {
      pf[e]     = (_Float16)(hiHalf ? p1s[e] : p0[e]);
      pf[e + 8] = (_Float16)(hiHalf ? p1[e] : p0s[e]);
    }

    // O^T += V^T(64 x 32keys) @ P^T(32keys x 16q): 4 d-tiles of 16
#pragma unroll
    for (int t = 0; t < 4; ++t) {
      v16h va = load_a_frag(&Vs[buf][0], 32, 16 * t, 0, lane);
      o_acc[t] = wmma_f16(va, pf, o_acc[t]);
    }
    __syncthreads();
  }

  // epilogue: normalize, write attn f16 in [B, N, E] row-major (E = h*64+d)
  const float inv = 1.0f / lsum;
  const int q = qbase + (lane & 15);
  const int dbase = (lane >> 4) << 3;
#pragma unroll
  for (int t = 0; t < 4; ++t) {
    v8h oh;
#pragma unroll
    for (int r = 0; r < 8; ++r) oh[r] = (_Float16)(o_acc[t][r] * inv);
    const int d = 16 * t + dbase;
    const size_t idx = ((size_t)(bi * SEQ + q)) * EMBED + h * HDIM + d;
    *(v8h*)(attn + idx) = oh;   // 8 contiguous d's -> one b128 store
  }
}

// ---------------------------------------------------------------------------
extern "C" void kernel_launch(void* const* d_in, const int* in_sizes, int n_in,
                              void* d_out, int out_size, void* d_ws, size_t ws_size,
                              hipStream_t stream) {
  const float* x     = (const float*)d_in[0];   // [B,N,E]
  const float* qkv_w = (const float*)d_in[1];   // [3E,E]
  const float* qkv_b = (const float*)d_in[2];   // [3E]
  const float* out_w = (const float*)d_in[3];   // [E,E]
  const float* out_b = (const float*)d_in[4];   // [E]
  float* out = (float*)d_out;                   // [B,N,E] fp32

  char* ws = (char*)d_ws;
  size_t off = 0;
  _Float16* xh   = (_Float16*)(ws + off); off += (size_t)ROWS * EMBED * 2;
  _Float16* wh1  = (_Float16*)(ws + off); off += (size_t)3 * EMBED * EMBED * 2;
  _Float16* wh2  = (_Float16*)(ws + off); off += (size_t)EMBED * EMBED * 2;
  _Float16* qh   = (_Float16*)(ws + off); off += (size_t)ROWS * EMBED * 2;
  _Float16* kh   = (_Float16*)(ws + off); off += (size_t)ROWS * EMBED * 2;
  _Float16* vth  = (_Float16*)(ws + off); off += (size_t)ROWS * EMBED * 2;
  _Float16* attn = (_Float16*)(ws + off); off += (size_t)ROWS * EMBED * 2;

  // 1) fp32 -> f16 conversions
  {
    int n4 = ROWS * EMBED / 4;
    cvt_f32_f16<<<(n4 + 255) / 256, 256, 0, stream>>>(x, xh, n4);
    n4 = 3 * EMBED * EMBED / 4;
    cvt_f32_f16<<<(n4 + 255) / 256, 256, 0, stream>>>(qkv_w, wh1, n4);
    n4 = EMBED * EMBED / 4;
    cvt_f32_f16<<<(n4 + 255) / 256, 256, 0, stream>>>(out_w, wh2, n4);
  }

  // 2) QKV projection: [8192,3072] = xh @ wh1^T + qkv_b, scatter Q/K/Vt f16
  gemm_f16<1><<<dim3(3 * EMBED / 256, ROWS / 64), 256, 0, stream>>>(
      xh, wh1, qkv_b, nullptr, qh, kh, vth, 3 * EMBED, EMBED);

  // 3) flash attention -> attn f16 [B,N,E]
  attention_f16<<<BATCH * HEADS * (SEQ / 16) / 8, 256, 0, stream>>>(
      qh, kh, vth, attn);

  // 4) output projection: out = attn @ wh2^T + out_b  (fp32)
  gemm_f16<0><<<dim3(EMBED / 256, ROWS / 64), 256, 0, stream>>>(
      attn, wh2, out_b, out, nullptr, nullptr, nullptr, EMBED, EMBED);
}
